// CausalSelfAttention_13142599926188
// MI455X (gfx1250) — compile-verified, matrix operand order checked by reading, not run on hardware
//
#include <hip/hip_runtime.h>

// Problem constants (match reference)
#define B_  2
#define S_  2048
#define D_  2048
#define H_  16
#define HD_ 128
#define M_  (B_ * S_)   // 4096 rows of x

typedef __bf16 v16bf __attribute__((ext_vector_type(16)));
typedef __bf16 v8bf  __attribute__((ext_vector_type(8)));
typedef __bf16 v4bf  __attribute__((ext_vector_type(4)));
typedef float  v8f   __attribute__((ext_vector_type(8)));
typedef float  v4f   __attribute__((ext_vector_type(4)));

// Load one 16x32 bf16 WMMA fragment row (A-layout) from LDS.
// rp points at the k-base of this lane's row; lo = (lane>>4)*8.
// Elements 0..7  = k in [lo, lo+8)        (VGPR0-3)
// Elements 8..15 = k in [16+lo, 16+lo+8)  (VGPR4-7)
static __device__ __forceinline__ v16bf frag_ld(const __bf16* rp, int lo) {
  v8bf a0 = *(const v8bf*)(rp + lo);
  v8bf a1 = *(const v8bf*)(rp + 16 + lo);
  return __builtin_shufflevector(a0, a1, 0, 1, 2, 3, 4, 5, 6, 7,
                                         8, 9, 10, 11, 12, 13, 14, 15);
}

static __device__ __forceinline__ v8f wmma_bf16(v16bf a, v16bf b, v8f c) {
  return __builtin_amdgcn_wmma_f32_16x16x32_bf16(false, a, false, b,
                                                 (short)0, c, false, false);
}

static __device__ __forceinline__ v8f zero8() {
  v8f z = {0.f, 0.f, 0.f, 0.f, 0.f, 0.f, 0.f, 0.f};
  return z;
}

// CDNA5 async copy: global -> LDS, 16 bytes per lane, tracked by ASYNCcnt.
// Low 32 bits of a flat LDS pointer are the wave-relative LDS byte address
static __device__ __forceinline__ void async_copy_b128(const __bf16* gsrc,
                                                       void* ldst) {
  unsigned loff = (unsigned)(unsigned long long)ldst;
  asm volatile("global_load_async_to_lds_b128 %0, %1, off"
               :
               : "v"(loff), "v"(gsrc)
               : "memory");
}
static __device__ __forceinline__ void async_wait0() {
  asm volatile("s_wait_asynccnt 0x0" ::: "memory");
}

// ---------------------------------------------------------------------------
// Kernel 1: fused projection + bias + (optional) RoPE, fp32 in -> bf16 out
// out layout: [B, H, S, HD] head-major.
// Block = 256 threads (8 waves). Tile: 128x128, BK = 32.
// Dynamic LDS: 65536 B ( A[128][40]bf16 | B[128][40]bf16, overlaid by C[128][128]f32 ).
// ---------------------------------------------------------------------------
__global__ __launch_bounds__(256)
void qkv_proj_rope(const float* __restrict__ X, const float* __restrict__ W,
                   const float* __restrict__ bias, __bf16* __restrict__ dst,
                   int do_rope) {
  extern __shared__ char smem[];
  __bf16(*As)[40] = (__bf16(*)[40])(smem);
  __bf16(*Bs)[40] = (__bf16(*)[40])(smem + 10240);
  float (*Cs)[128] = (float(*)[128])(smem);  // overlaid; used after final barrier

  const int tid  = threadIdx.x;
  const int lane = tid & 31;
  const int wid  = tid >> 5;
  const int wm   = wid & 3;        // wave row-group (0..3)  -> 32 rows
  const int wn   = wid >> 2;       // wave col-group (0..1)  -> 64 cols
  const int lm   = lane & 15;
  const int lo   = (lane >> 4) << 3;
  const int m0   = blockIdx.x * 128;
  const int n0   = blockIdx.y * 128;

  v8f acc[2][4];
#pragma unroll
  for (int i = 0; i < 2; ++i)
#pragma unroll
    for (int j = 0; j < 4; ++j) acc[i][j] = zero8();

  for (int k0 = 0; k0 < D_; k0 += 32) {
    // A tile: 128x32 fp32 -> bf16 LDS. 1024 chunks of 4 floats (b128 loads,
    // packed ds_store_b64).
#pragma unroll
    for (int t = 0; t < 4; ++t) {
      int ch = tid + t * 256;
      int r = ch >> 3, co = (ch & 7) * 4;
      v4f v = *(const v4f*)(X + (size_t)(m0 + r) * D_ + k0 + co);
      v4bf pk;
      pk[0] = (__bf16)v[0]; pk[1] = (__bf16)v[1];
      pk[2] = (__bf16)v[2]; pk[3] = (__bf16)v[3];
      *(v4bf*)&As[r][co] = pk;
    }
    // B tile: W[k][n] -> Bs[n][k] (n-major, k-contiguous). Coalesced b128
    // global loads along n, bf16 scatter into LDS.
#pragma unroll
    for (int t = 0; t < 4; ++t) {
      int ch = tid + t * 256;
      int k = ch >> 5, nc = (ch & 31) * 4;
      v4f wv = *(const v4f*)(W + (size_t)(k0 + k) * D_ + n0 + nc);
      Bs[nc + 0][k] = (__bf16)wv[0];
      Bs[nc + 1][k] = (__bf16)wv[1];
      Bs[nc + 2][k] = (__bf16)wv[2];
      Bs[nc + 3][k] = (__bf16)wv[3];
    }
    if (k0 + 32 < D_) {
      __builtin_prefetch(X + (size_t)(m0 + (tid >> 1)) * D_ + k0 + 32, 0, 1);
      __builtin_prefetch(W + (size_t)(k0 + 32 + (tid & 31)) * D_ + n0, 0, 1);
    }
    __syncthreads();

    v16bf af[2], bfm[4];
#pragma unroll
    for (int mt = 0; mt < 2; ++mt)
      af[mt] = frag_ld(&As[wm * 32 + mt * 16 + lm][0], lo);
#pragma unroll
    for (int nt = 0; nt < 4; ++nt)
      bfm[nt] = frag_ld(&Bs[wn * 64 + nt * 16 + lm][0], lo);
#pragma unroll
    for (int mt = 0; mt < 2; ++mt)
#pragma unroll
      for (int nt = 0; nt < 4; ++nt)
        acc[mt][nt] = wmma_bf16(af[mt], bfm[nt], acc[mt][nt]);
    __syncthreads();
  }

  // Spill accumulators to LDS (C layout: vgpr i -> row i (+8 for hi lanes))
#pragma unroll
  for (int mt = 0; mt < 2; ++mt)
#pragma unroll
    for (int nt = 0; nt < 4; ++nt)
#pragma unroll
      for (int i = 0; i < 8; ++i)
        Cs[wm * 32 + mt * 16 + i + lo][wn * 64 + nt * 16 + lm] = acc[mt][nt][i];
  __syncthreads();

  // bias + RoPE + store bf16 to [B,H,S,HD]; this block spans exactly head n0>>7.
  const int h = n0 >> 7;
  for (int e = tid; e < 128 * 64; e += 256) {
    int r = e >> 6, c = e & 63;
    float v1 = Cs[r][c] + bias[n0 + c];
    float v2 = Cs[r][c + 64] + bias[n0 + c + 64];
    int gm = m0 + r;
    int bi = gm >> 11;          // / S_
    int s  = gm & (S_ - 1);
    float o1 = v1, o2 = v2;
    if (do_rope) {
      // inv_freq = 10000^(-2c/HD) = exp(-(ln 1e4 / 64) * c)
      float ang = (float)s * __expf(-0.14391157f * (float)c);
      float sn, cs;
      __sincosf(ang, &sn, &cs);
      o1 = v1 * cs - v2 * sn;
      o2 = v1 * sn + v2 * cs;
    }
    __bf16* p = dst + (((size_t)bi * H_ + h) * S_ + s) * HD_;
    p[c]      = (__bf16)o1;
    p[c + 64] = (__bf16)o2;
  }
}

// ---------------------------------------------------------------------------
// Kernel 2: flash attention per (b,h, 64-row q tile). 256 threads (8 waves).
// Q and K tiles staged with global_load_async_to_lds_b128 (ASYNCcnt path).
// Dynamic LDS layout (bytes):
//   Qs [64][136] bf16  @ 0       (17408)
//   Ks [64][136] bf16  @ 17408   (17408)
//   Vs [128][72] bf16  @ 34816   (18432)   (V transposed: [d][kv])
//   Ss [64][68]  f32   @ 53248   (17408)
//   Ps [64][72]  bf16  @ 70656   ( 9216)
//   alpha[64] f32      @ 79872   (  256)
//   linv [64] f32      @ 80128   (  256)   total 80384
// ---------------------------------------------------------------------------
__global__ __launch_bounds__(256)
void attn_fwd(const __bf16* __restrict__ Q, const __bf16* __restrict__ K,
              const __bf16* __restrict__ V, __bf16* __restrict__ O) {
  extern __shared__ char smem[];
  __bf16(*Qs)[136] = (__bf16(*)[136])(smem);
  __bf16(*Ks)[136] = (__bf16(*)[136])(smem + 17408);
  __bf16(*Vs)[72]  = (__bf16(*)[72])(smem + 34816);
  float (*Ss)[68]  = (float(*)[68])(smem + 53248);
  __bf16(*Ps)[72]  = (__bf16(*)[72])(smem + 70656);
  float* alphaS    = (float*)(smem + 79872);
  float* linvS     = (float*)(smem + 80128);

  const int tid  = threadIdx.x;
  const int lane = tid & 31;
  const int wid  = tid >> 5;
  const int lm   = lane & 15;
  const int lo   = (lane >> 4) << 3;
  const int mt   = wid & 3;    // q row-group of this wave (16 rows)
  const int ng   = wid >> 2;   // col half (0/1)

  const int bh = blockIdx.y;   // b*H + h
  const int b  = bh >> 4;
  const int h  = bh & 15;
  const int q0 = blockIdx.x * 64;

  // Q tile: async global->LDS copy (16B per lane per chunk)
  const __bf16* Qp = Q + ((size_t)bh * S_ + q0) * HD_;
#pragma unroll
  for (int t = 0; t < 4; ++t) {
    int ch = tid + t * 256;
    int r = ch >> 4, co = (ch & 15) * 8;
    async_copy_b128(Qp + (size_t)r * HD_ + co, &Qs[r][co]);
  }

  v8f oacc[4];
#pragma unroll
  for (int i = 0; i < 4; ++i) oacc[i] = zero8();
  float m_r = -1e30f, l_r = 0.f;
  const float scale = 0.08838834764831845f;  // 1/sqrt(128)

  for (int j = 0; j <= (int)blockIdx.x; ++j) {
    const int k0 = j * 64;
    const __bf16* Kp = K + ((size_t)bh * S_ + k0) * HD_;
    const __bf16* Vp = V + ((size_t)bh * S_ + k0) * HD_;
    // K tile: async copy; V tile: load to VGPR and transpose-scatter to LDS.
#pragma unroll
    for (int t = 0; t < 4; ++t) {
      int ch = tid + t * 256;
      int r = ch >> 4, co = (ch & 15) * 8;
      async_copy_b128(Kp + (size_t)r * HD_ + co, &Ks[r][co]);
      v8bf vv = *(const v8bf*)(Vp + (size_t)r * HD_ + co);
#pragma unroll
      for (int i = 0; i < 8; ++i) Vs[co + i][r] = vv[i];
    }
    async_wait0();      // ASYNCcnt -> 0: Q (first iter) + K tile landed in LDS
    __syncthreads();

    // ---- S = scale * Q @ K^T  (64x64), each wave: 1 m-tile x 2 n-tiles ----
    v8f sacc[2];
    sacc[0] = zero8();
    sacc[1] = zero8();
#pragma unroll
    for (int kd = 0; kd < 4; ++kd) {
      v16bf a = frag_ld(&Qs[mt * 16 + lm][kd * 32], lo);
#pragma unroll
      for (int p = 0; p < 2; ++p) {
        v16bf bb = frag_ld(&Ks[(ng * 2 + p) * 16 + lm][kd * 32], lo);
        sacc[p] = wmma_bf16(a, bb, sacc[p]);
      }
    }
#pragma unroll
    for (int p = 0; p < 2; ++p)
#pragma unroll
      for (int i = 0; i < 8; ++i)
        Ss[mt * 16 + i + lo][(ng * 2 + p) * 16 + lm] = sacc[p][i] * scale;
    __syncthreads();

    // ---- online softmax: one thread per q row ----
    if (tid < 64) {
      const int r = tid, qg = q0 + r;
      float mx = -1e30f;
      for (int c = 0; c < 64; ++c) {
        if (k0 + c > qg) break;  // causal: columns increase
        mx = fmaxf(mx, Ss[r][c]);
      }
      float mnew  = fmaxf(m_r, mx);
      float alpha = __expf(m_r - mnew);
      float ls = 0.f;
      for (int c = 0; c < 64; ++c) {
        float pv = 0.f;
        if (k0 + c <= qg) pv = __expf(Ss[r][c] - mnew);
        Ps[r][c] = (__bf16)pv;
        ls += pv;
      }
      l_r = l_r * alpha + ls;
      m_r = mnew;
      alphaS[r] = alpha;
    }
    __syncthreads();

    // ---- O = O*alpha + P @ V ; each wave: 1 m-tile x 4 n-tiles of 16 ----
    float al[8];
#pragma unroll
    for (int i = 0; i < 8; ++i) al[i] = alphaS[mt * 16 + i + lo];
#pragma unroll
    for (int nt = 0; nt < 4; ++nt)
#pragma unroll
      for (int i = 0; i < 8; ++i) oacc[nt][i] *= al[i];
#pragma unroll
    for (int kd = 0; kd < 2; ++kd) {
      v16bf a = frag_ld(&Ps[mt * 16 + lm][kd * 32], lo);
#pragma unroll
      for (int nt = 0; nt < 4; ++nt) {
        v16bf bb = frag_ld(&Vs[(ng * 4 + nt) * 16 + lm][kd * 32], lo);
        oacc[nt] = wmma_bf16(a, bb, oacc[nt]);
      }
    }
    __syncthreads();
  }

  if (tid < 64) linvS[tid] = 1.f / l_r;
  __syncthreads();
  float li[8];
#pragma unroll
  for (int i = 0; i < 8; ++i) li[i] = linvS[mt * 16 + i + lo];
#pragma unroll
  for (int nt = 0; nt < 4; ++nt) {
    const int d = (ng * 4 + nt) * 16 + lm;
#pragma unroll
    for (int i = 0; i < 8; ++i) {
      const int row = q0 + mt * 16 + i + lo;
      O[((size_t)b * S_ + row) * D_ + h * HD_ + d] = (__bf16)(oacc[nt][i] * li[i]);
    }
  }
}

// ---------------------------------------------------------------------------
// Kernel 3: output projection: bf16 A [M_, D_] @ fp32 W + bias -> fp32 out
// Dynamic LDS: 20480 B (A[128][40] | B[128][40] bf16)
// ---------------------------------------------------------------------------
__global__ __launch_bounds__(256)
void out_proj(const __bf16* __restrict__ A, const float* __restrict__ W,
              const float* __restrict__ bias, float* __restrict__ out) {
  extern __shared__ char smem[];
  __bf16(*As)[40] = (__bf16(*)[40])(smem);
  __bf16(*Bs)[40] = (__bf16(*)[40])(smem + 10240);

  const int tid  = threadIdx.x;
  const int lane = tid & 31;
  const int wid  = tid >> 5;
  const int wm   = wid & 3;
  const int wn   = wid >> 2;
  const int lm   = lane & 15;
  const int lo   = (lane >> 4) << 3;
  const int m0   = blockIdx.x * 128;
  const int n0   = blockIdx.y * 128;

  v8f acc[2][4];
#pragma unroll
  for (int i = 0; i < 2; ++i)
#pragma unroll
    for (int j = 0; j < 4; ++j) acc[i][j] = zero8();

  for (int k0 = 0; k0 < D_; k0 += 32) {
    // A tile: 128x32 bf16, 512 chunks of 8 (16B loads)
#pragma unroll
    for (int t = 0; t < 2; ++t) {
      int ch = tid + t * 256;
      int r = ch >> 2, co = (ch & 3) * 8;
      *(v8bf*)&As[r][co] = *(const v8bf*)(A + (size_t)(m0 + r) * D_ + k0 + co);
    }
    // B tile: coalesced b128 loads along n, bf16 scatter into LDS.
#pragma unroll
    for (int t = 0; t < 4; ++t) {
      int ch = tid + t * 256;
      int k = ch >> 5, nc = (ch & 31) * 4;
      v4f wv = *(const v4f*)(W + (size_t)(k0 + k) * D_ + n0 + nc);
      Bs[nc + 0][k] = (__bf16)wv[0];
      Bs[nc + 1][k] = (__bf16)wv[1];
      Bs[nc + 2][k] = (__bf16)wv[2];
      Bs[nc + 3][k] = (__bf16)wv[3];
    }
    if (k0 + 32 < D_) {
      __builtin_prefetch(A + (size_t)(m0 + (tid >> 1)) * D_ + k0 + 32, 0, 1);
      __builtin_prefetch(W + (size_t)(k0 + 32 + (tid & 31)) * D_ + n0, 0, 1);
    }
    __syncthreads();

    v16bf af[2], bfm[4];
#pragma unroll
    for (int mt = 0; mt < 2; ++mt)
      af[mt] = frag_ld(&As[wm * 32 + mt * 16 + lm][0], lo);
#pragma unroll
    for (int nt = 0; nt < 4; ++nt)
      bfm[nt] = frag_ld(&Bs[wn * 64 + nt * 16 + lm][0], lo);
#pragma unroll
    for (int mt = 0; mt < 2; ++mt)
#pragma unroll
      for (int nt = 0; nt < 4; ++nt)
        acc[mt][nt] = wmma_bf16(af[mt], bfm[nt], acc[mt][nt]);
    __syncthreads();
  }

#pragma unroll
  for (int mt = 0; mt < 2; ++mt)
#pragma unroll
    for (int nt = 0; nt < 4; ++nt) {
      const int col = n0 + wn * 64 + nt * 16 + lm;
      const float bb = bias[col];
#pragma unroll
      for (int i = 0; i < 8; ++i) {
        const int row = m0 + wm * 32 + mt * 16 + i + lo;
        out[(size_t)row * D_ + col] = acc[mt][nt][i] + bb;
      }
    }
}

// ---------------------------------------------------------------------------
extern "C" void kernel_launch(void* const* d_in, const int* in_sizes, int n_in,
                              void* d_out, int out_size, void* d_ws,
                              size_t ws_size, hipStream_t stream) {
  (void)in_sizes; (void)n_in; (void)out_size; (void)ws_size;
  // inputs: x, attn_mask, Wq, bq, Wk, bk, Wv, bv, Wo, bo
  const float* x  = (const float*)d_in[0];
  const float* Wq = (const float*)d_in[2];
  const float* bq = (const float*)d_in[3];
  const float* Wk = (const float*)d_in[4];
  const float* bk = (const float*)d_in[5];
  const float* Wv = (const float*)d_in[6];
  const float* bv = (const float*)d_in[7];
  const float* Wo = (const float*)d_in[8];
  const float* bo = (const float*)d_in[9];
  float* out = (float*)d_out;

  const size_t per = (size_t)B_ * H_ * S_ * HD_;  // elements per Q/K/V tensor
  __bf16* qws = (__bf16*)d_ws;
  __bf16* kws = qws + per;
  __bf16* vws = kws + per;
  __bf16* aws = vws + per;  // attention output, [B,S,D] bf16

  dim3 blk(256);
  dim3 gproj(M_ / 128, D_ / 128);  // 32 x 16
  qkv_proj_rope<<<gproj, blk, 65536, stream>>>(x, Wq, bq, qws, 1);
  qkv_proj_rope<<<gproj, blk, 65536, stream>>>(x, Wk, bk, kws, 1);
  qkv_proj_rope<<<gproj, blk, 65536, stream>>>(x, Wv, bv, vws, 0);
  attn_fwd<<<dim3(S_ / 64, B_ * H_), blk, 80384, stream>>>(qws, kws, vws, aws);
  out_proj<<<gproj, blk, 20480, stream>>>(aws, Wo, bo, out);
}
